// VectorQuantizer_60550448939194
// MI455X (gfx1250) — matrix operand order repaired and checked by reading, not compiled, not run on hardware
//
#include <hip/hip_runtime.h>
#include <stdint.h>

// ---------------------------------------------------------------------------
// VQ-VAE vector quantizer for MI455X (gfx1250, wave32, WMMA).
//   score(n,k) = z_n . e_k - 0.5*||e_k||^2 ; argmax_k == argmin_k distance
//   GEMM done as bf16x3 split (hi/lo) with f32 WMMA accumulation.
//   E tiles staged to LDS via GLOBAL_LOAD_ASYNC_TO_LDS_B128 (ASYNCcnt).
// ---------------------------------------------------------------------------

typedef __attribute__((ext_vector_type(16))) __bf16 v16bf;
typedef __attribute__((ext_vector_type(8)))  float  v8f;
typedef int v4i __attribute__((vector_size(16)));   // matches builtin param type

union Frag { v16bf v; uint4 q[2]; };

constexpr int N_TOK = 32768;
constexpr int N_EMB = 8192;
constexpr int DIM   = 256;

// workspace layout (bytes)
constexpr size_t WS_EHI   = 0;                                  // N_EMB*DIM bf16
constexpr size_t WS_ELO   = WS_EHI   + (size_t)N_EMB * DIM * 2; // N_EMB*DIM bf16
constexpr size_t WS_ZHI   = WS_ELO   + (size_t)N_EMB * DIM * 2; // N_TOK*DIM bf16
constexpr size_t WS_ZLO   = WS_ZHI   + (size_t)N_TOK * DIM * 2; // N_TOK*DIM bf16
constexpr size_t WS_EBIAS = WS_ZLO   + (size_t)N_TOK * DIM * 2; // N_EMB f32 (-0.5*||e||^2)
constexpr size_t WS_IDX   = WS_EBIAS + (size_t)N_EMB * 4;       // N_TOK i32
constexpr size_t WS_PART  = WS_IDX   + (size_t)N_TOK * 4;       // 4096 f32

#define HAS_ASYNC_LDS  __has_builtin(__builtin_amdgcn_global_load_async_to_lds_b128)

__device__ __forceinline__ unsigned short f2bf(float f) {
    union { float f; unsigned u; } x; x.f = f;
    unsigned u = x.u;
    unsigned r = u + 0x7FFFu + ((u >> 16) & 1u);   // round-to-nearest-even
    return (unsigned short)(r >> 16);
}
__device__ __forceinline__ float bf2f(unsigned short h) {
    union { unsigned u; float f; } x; x.u = ((unsigned)h) << 16;
    return x.f;
}

__device__ __forceinline__ v8f wmma_bf16(v16bf a, v16bf b, v8f c) {
    return __builtin_amdgcn_wmma_f32_16x16x32_bf16(false, a, false, b,
                                                   (short)0, c, false, false);
}

#if HAS_ASYNC_LDS
typedef __attribute__((address_space(1))) v4i g_v4i;
typedef __attribute__((address_space(3))) v4i l_v4i;

// 32B memory -> LDS, two async b128s
__device__ __forceinline__ void async_copy32(const void* gsrc, void* ldst) {
    g_v4i* g = (g_v4i*)(uintptr_t)gsrc;
    l_v4i* l = (l_v4i*)(uint32_t)(uintptr_t)ldst;  // flat LDS addr low bits = LDS offset
    __builtin_amdgcn_global_load_async_to_lds_b128(g,     l,     0, 0);
    __builtin_amdgcn_global_load_async_to_lds_b128(g + 1, l + 1, 0, 0);
}
#endif

template <int N>
__device__ __forceinline__ void wait_async() {
#if __has_builtin(__builtin_amdgcn_s_wait_asynccnt)
    __builtin_amdgcn_s_wait_asynccnt(N);
#else
    asm volatile("s_wait_asynccnt %0" :: "i"(N) : "memory");
#endif
}

// ---------------------------------------------------------------------------
// Kernel 1: split embeddings into bf16 hi/lo, compute bias = -0.5*||e||^2
// ---------------------------------------------------------------------------
__global__ void vq_prep_embed(const float* __restrict__ E,
                              unsigned short* __restrict__ eh,
                              unsigned short* __restrict__ el,
                              float* __restrict__ ebias) {
    const int k = blockIdx.x;
    const int d = threadIdx.x;
    const float v = E[(size_t)k * DIM + d];
    const unsigned short h = f2bf(v);
    const float hf = bf2f(h);
    eh[(size_t)k * DIM + d] = h;
    el[(size_t)k * DIM + d] = f2bf(v - hf);

    __shared__ float red[256];
    red[d] = v * v;
    __syncthreads();
    for (int s = 128; s > 0; s >>= 1) {
        if (d < s) red[d] += red[d + s];
        __syncthreads();
    }
    if (d == 0) ebias[k] = -0.5f * red[0];
}

// ---------------------------------------------------------------------------
// Kernel 2: split z_e into bf16 hi/lo (elementwise)
// ---------------------------------------------------------------------------
__global__ void vq_prep_z(const float* __restrict__ Z,
                          unsigned short* __restrict__ zh,
                          unsigned short* __restrict__ zl) {
    const size_t i = (size_t)blockIdx.x * 256 + threadIdx.x;
    const float v = Z[i];
    const unsigned short h = f2bf(v);
    zh[i] = h;
    zl[i] = f2bf(v - bf2f(h));
}

// ---------------------------------------------------------------------------
// Kernel 3: fused GEMM + per-row argmax.
//   256 blocks x 256 threads (8 waves). Block handles 128 token rows;
//   wave handles 16 rows x all 8192 codes. E tiles double-buffered in LDS.
// ---------------------------------------------------------------------------
__global__ __launch_bounds__(256, 1) void vq_argmin(
    const unsigned short* __restrict__ zh, const unsigned short* __restrict__ zl,
    const unsigned short* __restrict__ eh, const unsigned short* __restrict__ el,
    const float* __restrict__ ebias, int* __restrict__ outIdx) {

    constexpr int SPB = 528;  // padded LDS row stride (bytes) = 33*16 -> bank spread
    __shared__ __align__(16) unsigned char lds[2][2][16 * SPB];  // [buf][hi/lo]

    const int tid  = threadIdx.x;
    const int lane = tid & 31;
    const int wave = tid >> 5;
    const int h    = lane >> 4;   // half-wave
    const int n    = lane & 15;   // tile column (codes) / A row (tokens)
    const int R    = blockIdx.x * 128 + wave * 16;

    // --- load A fragments (16 rows x 256 dims, hi & lo) into registers ---
    // A layout used here (16-bit 16x32, contiguous k): lane holds row m=lane%16;
    //   elems 0..7  -> K = 32c + 16h + {0..7}
    //   elems 8..15 -> K = 32c + 16h + 8 + {0..7}
    Frag ahi[8], alo[8];
    {
        const unsigned short* zrh = zh + (size_t)(R + n) * DIM;
        const unsigned short* zrl = zl + (size_t)(R + n) * DIM;
#pragma unroll
        for (int c = 0; c < 8; ++c) {
            const int k0 = 32 * c + 16 * h;
            ahi[c].q[0] = *(const uint4*)(zrh + k0);
            ahi[c].q[1] = *(const uint4*)(zrh + k0 + 8);
            alo[c].q[0] = *(const uint4*)(zrl + k0);
            alo[c].q[1] = *(const uint4*)(zrl + k0 + 8);
        }
    }

    // staging: each thread moves 32B of hi + 32B of lo per tile (tile = 8KB+8KB)
    const int f    = tid * 32;
    const int srow = f >> 9;     // f / 512
    const int scol = f & 511;
    const unsigned char* gH = (const unsigned char*)eh + f;  // + t*8192
    const unsigned char* gL = (const unsigned char*)el + f;
    unsigned char* dH[2] = { &lds[0][0][srow * SPB + scol], &lds[1][0][srow * SPB + scol] };
    unsigned char* dL[2] = { &lds[0][1][srow * SPB + scol], &lds[1][1][srow * SPB + scol] };

    const int NTILE = N_EMB / 16;  // 512

#if HAS_ASYNC_LDS
    // prologue: async-stage tile 0 into buffer 0
    async_copy32(gH, dH[0]);
    async_copy32(gL, dL[0]);
#else
    uint4 rh0, rh1, rl0, rl1;
    rh0 = *(const uint4*)gH; rh1 = *(const uint4*)(gH + 16);
    rl0 = *(const uint4*)gL; rl1 = *(const uint4*)(gL + 16);
#endif

    float runMax[8];
    int   runIdx[8];
#pragma unroll
    for (int j = 0; j < 8; ++j) { runMax[j] = -3.4e38f; runIdx[j] = 0; }

    float biasCur = ebias[n];  // tile 0 bias, prefetched ahead each iteration

    for (int t = 0; t < NTILE; ++t) {
        const int buf = t & 1;

#if HAS_ASYNC_LDS
        if (t + 1 < NTILE) {
            async_copy32(gH + (size_t)(t + 1) * 8192, dH[buf ^ 1]);
            async_copy32(gL + (size_t)(t + 1) * 8192, dL[buf ^ 1]);
            wait_async<4>();   // tile t done; only tile t+1's 4 loads in flight
        } else {
            wait_async<0>();
        }
#else
        // commit staged tile to LDS, prefetch next into registers
        *(uint4*)dH[buf] = rh0; *(uint4*)(dH[buf] + 16) = rh1;
        *(uint4*)dL[buf] = rl0; *(uint4*)(dL[buf] + 16) = rl1;
        if (t + 1 < NTILE) {
            const unsigned char* ph = gH + (size_t)(t + 1) * 8192;
            const unsigned char* pl = gL + (size_t)(t + 1) * 8192;
            rh0 = *(const uint4*)ph; rh1 = *(const uint4*)(ph + 16);
            rl0 = *(const uint4*)pl; rl1 = *(const uint4*)(pl + 16);
        }
#endif
        // prefetch next tile's bias (latency hidden behind this tile's compute)
        const float biasNext = (t + 1 < NTILE) ? ebias[(t + 1) * 16 + n] : 0.0f;

        __syncthreads();

        v8f acc0 = {}, acc1 = {};   // literal-0 C operand for first WMMA of each chain

        // B layout used here (16-bit 32x16, same contiguous k as A): lane holds col n=lane%16,
        //   elems 0..15 -> K = 32c + 16h + {0..15}  (contiguous 32B in the row)
        const unsigned char* bhp = &lds[buf][0][n * SPB + 32 * h];
        const unsigned char* blp = &lds[buf][1][n * SPB + 32 * h];
#pragma unroll
        for (int c = 0; c < 8; ++c) {
            Frag bhi, blo;
            bhi.q[0] = *(const uint4*)(bhp + 64 * c);
            bhi.q[1] = *(const uint4*)(bhp + 64 * c + 16);
            blo.q[0] = *(const uint4*)(blp + 64 * c);
            blo.q[1] = *(const uint4*)(blp + 64 * c + 16);
            v8f& acc = (c & 1) ? acc1 : acc0;
            acc = wmma_bf16(ahi[c].v, bhi.v, acc);  // hi*hi
            acc = wmma_bf16(ahi[c].v, blo.v, acc);  // hi*lo
            acc = wmma_bf16(alo[c].v, bhi.v, acc);  // lo*hi
        }

        const int idx = t * 16 + n;
#pragma unroll
        for (int j = 0; j < 8; ++j) {
            const float s = acc0[j] + acc1[j] + biasCur;
            const bool gt = s > runMax[j];          // strict: keeps first index
            runMax[j] = gt ? s : runMax[j];
            runIdx[j] = gt ? idx : runIdx[j];
        }
        biasCur = biasNext;
    }

    // cross-lane argmax over the 16 column-lanes of each half-wave
#pragma unroll
    for (int j = 0; j < 8; ++j) {
        float m  = runMax[j];
        int   id = runIdx[j];
#pragma unroll
        for (int off = 1; off < 16; off <<= 1) {
            const float om = __shfl_xor(m, off, 32);
            const int   oi = __shfl_xor(id, off, 32);
            if (om > m || (om == m && oi < id)) { m = om; id = oi; }
        }
        if (n == 0) outIdx[R + 8 * h + j] = id;  // C row M = j + 8h
    }
}

// ---------------------------------------------------------------------------
// Kernel 4: gather z_q -> out, squared-error partial sums, indices as f32
// ---------------------------------------------------------------------------
__global__ void vq_gather(const float* __restrict__ Z, const float* __restrict__ E,
                          const int* __restrict__ idx, float* __restrict__ out,
                          float* __restrict__ part) {
    const int tid  = threadIdx.x;
    const int tok  = blockIdx.x * 8 + (tid >> 5);
    const int lane = tid & 31;
    const int k    = idx[tok];

    const float4* zr = (const float4*)(Z + (size_t)tok * DIM) + lane * 2;
    const float4* er = (const float4*)(E + (size_t)k   * DIM) + lane * 2;
    float4*       o  = (float4*)(out + (size_t)tok * DIM) + lane * 2;

    float s = 0.0f;
#pragma unroll
    for (int i = 0; i < 2; ++i) {
        const float4 z = zr[i];
        const float4 e = er[i];
        o[i] = e;  // z_q_st == z_q numerically
        const float dx = z.x - e.x, dy = z.y - e.y, dz = z.z - e.z, dw = z.w - e.w;
        s += dx * dx + dy * dy + dz * dz + dw * dw;
    }
    if (lane == 0) out[(size_t)N_TOK * DIM + 1 + tok] = (float)k;

    __shared__ float red[256];
    red[tid] = s;
    __syncthreads();
    for (int st = 128; st > 0; st >>= 1) {
        if (tid < st) red[tid] += red[tid + st];
        __syncthreads();
    }
    if (tid == 0) part[blockIdx.x] = red[0];
}

// ---------------------------------------------------------------------------
// Kernel 5: final loss reduction: vq_loss = 1.25 * mean((z-zq)^2)
// ---------------------------------------------------------------------------
__global__ void vq_finish(const float* __restrict__ part, float* __restrict__ out) {
    __shared__ float red[256];
    float s = 0.0f;
    for (int i = threadIdx.x; i < 4096; i += 256) s += part[i];
    red[threadIdx.x] = s;
    __syncthreads();
    for (int st = 128; st > 0; st >>= 1) {
        if (threadIdx.x < st) red[threadIdx.x] += red[threadIdx.x + st];
        __syncthreads();
    }
    if (threadIdx.x == 0)
        out[(size_t)N_TOK * DIM] = red[0] * (1.25f / ((float)N_TOK * (float)DIM));
}

// ---------------------------------------------------------------------------
extern "C" void kernel_launch(void* const* d_in, const int* in_sizes, int n_in,
                              void* d_out, int out_size, void* d_ws, size_t ws_size,
                              hipStream_t stream) {
    (void)in_sizes; (void)n_in; (void)out_size; (void)ws_size;

    const float* Z = (const float*)d_in[0];   // z_e       (32768, 256) f32
    const float* E = (const float*)d_in[1];   // embeddings (8192, 256) f32
    float* out = (float*)d_out;               // [z_q_st | vq_loss | indices] f32

    unsigned char* ws = (unsigned char*)d_ws;
    unsigned short* eh    = (unsigned short*)(ws + WS_EHI);
    unsigned short* el    = (unsigned short*)(ws + WS_ELO);
    unsigned short* zhp   = (unsigned short*)(ws + WS_ZHI);
    unsigned short* zlp   = (unsigned short*)(ws + WS_ZLO);
    float*          ebias = (float*)(ws + WS_EBIAS);
    int*            widx  = (int*)(ws + WS_IDX);
    float*          part  = (float*)(ws + WS_PART);

    vq_prep_embed<<<N_EMB, 256, 0, stream>>>(E, eh, el, ebias);
    vq_prep_z<<<(N_TOK * DIM) / 256, 256, 0, stream>>>(Z, zhp, zlp);
    vq_argmin<<<N_TOK / 128, 256, 0, stream>>>(zhp, zlp, eh, el, ebias, widx);
    vq_gather<<<N_TOK / 8, 256, 0, stream>>>(Z, E, widx, out, part);
    vq_finish<<<1, 256, 0, stream>>>(part, out);
}
